// ParallelMultiScaleHypergraphConv_67010079752630
// MI455X (gfx1250) — matrix operand order repaired and checked by reading, not compiled, not run on hardware
//
#include <hip/hip_runtime.h>
#include <hip/hip_bf16.h>
#include <stdint.h>

// ---- types for CDNA5 WMMA ----
typedef __bf16 bf16;
typedef bf16  v16bf __attribute__((ext_vector_type(16)));
typedef bf16  v8bf  __attribute__((ext_vector_type(8)));
typedef float v8f   __attribute__((ext_vector_type(8)));
typedef float v4f   __attribute__((ext_vector_type(4)));

namespace {

constexpr int NN  = 10000;  // N_NODES (625 * 16 exactly)
constexpr int NE  = 4000;   // N_EDGES
constexpr int KC  = 128;    // IN_CH
constexpr int HO  = 128;    // HEADS*OUT_CH
constexpr int HO2 = 256;    // both scales concatenated
constexpr int OC  = 64;     // OUT_CH

constexpr size_t al(size_t x) { return (x + 255) & ~size_t(255); }
// workspace layout (bytes)
constexpr size_t OFF_XBF  = 0;
constexpr size_t SZ_XBF   = al((size_t)NN * KC * 2);
constexpr size_t OFF_WCAT = OFF_XBF + SZ_XBF;          // [256,128] bf16 = [W1;W2]
constexpr size_t SZ_WCAT  = al((size_t)HO2 * KC * 2);
constexpr size_t OFF_BCAT = OFF_WCAT + SZ_WCAT;        // [256] f32 = [b1;b2]
constexpr size_t SZ_BCAT  = al((size_t)HO2 * 4);
constexpr size_t OFF_WOUT = OFF_BCAT + SZ_BCAT;        // [64,128] bf16
constexpr size_t SZ_WOUT  = al((size_t)OC * HO * 2);
constexpr size_t OFF_XT   = OFF_WOUT + SZ_WOUT;        // [10000,256] f32
constexpr size_t SZ_XT    = al((size_t)NN * HO2 * 4);
constexpr size_t OFF_ES1  = OFF_XT + SZ_XT;            // [4000,128] f32
constexpr size_t SZ_ES    = al((size_t)NE * HO * 4);
constexpr size_t OFF_CN1  = OFF_ES1 + SZ_ES;           // [4000] f32
constexpr size_t SZ_CN    = al((size_t)NE * 4);
constexpr size_t OFF_ES2  = OFF_CN1 + SZ_CN;
constexpr size_t OFF_CN2  = OFF_ES2 + SZ_ES;
constexpr size_t OFF_O    = OFF_CN2 + SZ_CN;           // [10000,128] f32 accum
constexpr size_t SZ_O     = al((size_t)NN * HO * 4);
constexpr size_t ZERO_FLOATS = (OFF_O + SZ_O - OFF_ES1) / 4;

} // namespace

// native fp32 global atomic add (no-return form, agent-visible through L2)
__device__ __forceinline__ void gatomic_add_f32(float* p, float v) {
    asm volatile("global_atomic_add_f32 %0, %1, off" :: "v"(p), "v"(v) : "memory");
}

// ---------------- conversion / init ----------------
__global__ __launch_bounds__(256) void k_convert(
    const float* __restrict__ x, const float* __restrict__ W1,
    const float* __restrict__ b1, const float* __restrict__ W2,
    const float* __restrict__ b2, const float* __restrict__ Wout,
    bf16* __restrict__ xb, bf16* __restrict__ wcat,
    float* __restrict__ bcat, bf16* __restrict__ wob) {
    const int NX = NN * KC;            // 1,280,000
    const int NW = HO2 * KC;           // 32,768
    int t = blockIdx.x * 256 + threadIdx.x;
    if (t < NX) {
        xb[t] = (bf16)x[t];
    } else if (t < NX + NW) {
        int u = t - NX;
        wcat[u] = (bf16)((u < HO * KC) ? W1[u] : W2[u - HO * KC]);
    } else if (t < NX + NW + HO2) {
        int v = t - NX - NW;
        bcat[v] = (v < HO) ? b1[v] : b2[v - HO];
    } else if (t < NX + NW + HO2 + OC * HO) {
        int w = t - NX - NW - HO2;
        wob[w] = (bf16)Wout[w];
    }
}

__global__ __launch_bounds__(256) void k_zero(float* __restrict__ p, size_t n) {
    size_t t = (size_t)blockIdx.x * 256 + threadIdx.x;
    if (t < n) p[t] = 0.0f;
}

// ---------------- GEMM1: xt[10000,256] = x_bf16 @ [W1;W2]^T + bcat ----------------
// one 16x16 tile per wave32; K loop = 4 x v_wmma_f32_16x16x32_bf16
__global__ __launch_bounds__(256) void k_gemm_xt(
    const bf16* __restrict__ xb, const bf16* __restrict__ w,
    const float* __restrict__ bias, float* __restrict__ xt) {
    const int tile = blockIdx.x * 8 + (threadIdx.x >> 5);
    if (tile >= (NN / 16) * (HO2 / 16)) return;        // wave-uniform guard
    const int lane = threadIdx.x & 31;
    const int mt = tile / (HO2 / 16);
    const int nt = tile % (HO2 / 16);
    const int m0 = mt * 16, n0 = nt * 16;
    const int h = lane >> 4;                            // wave half
    const int l = lane & 15;
    // A fragment: row m0+l; halves split K (h=0 -> K 0..7,16..23 ; h=1 -> +8)
    const bf16* arow = xb + (size_t)(m0 + l) * KC + h * 8;
    // B fragment: B[k][n] = Wcat[n][k]; same K split as A
    const bf16* brow = w + (size_t)(n0 + l) * KC + h * 8;
    v8f acc = {};
#pragma unroll
    for (int kb = 0; kb < KC; kb += 32) {
        v8bf alo = *(const v8bf*)(arow + kb);
        v8bf ahi = *(const v8bf*)(arow + kb + 16);
        v8bf bl0 = *(const v8bf*)(brow + kb);
        v8bf bl1 = *(const v8bf*)(brow + kb + 16);
        v16bf a, b;
#pragma unroll
        for (int e = 0; e < 8; ++e) {
            a[e] = alo[e]; a[8 + e] = ahi[e];
            b[e] = bl0[e]; b[8 + e] = bl1[e];
        }
        acc = __builtin_amdgcn_wmma_f32_16x16x32_bf16(
            false, a, false, b, (short)0, acc, false, false);
    }
    const float bv = bias[n0 + l];
#pragma unroll
    for (int r = 0; r < 8; ++r)   // C layout: VGPR r -> M = r (+8 for upper half)
        xt[(size_t)(m0 + h * 8 + r) * HO2 + n0 + l] = acc[r] + bv;
}

// ---------------- edge scatter: esum[e] += xt[node], cnt[e] += 1 ----------------
__global__ __launch_bounds__(256) void k_scatter_edge(
    const int* __restrict__ idx, int nnz, const float* __restrict__ xt,
    int colbase, float* __restrict__ esum, float* __restrict__ cnt) {
    size_t t = (size_t)blockIdx.x * 256 + threadIdx.x;
    if (t >= (size_t)nnz * HO) return;
    int i = (int)(t >> 7);          // / 128
    int c = (int)(t & 127);
    int node = idx[i];
    int edge = idx[nnz + i];
    gatomic_add_f32(esum + (size_t)edge * HO + c,
                    xt[(size_t)node * HO2 + colbase + c]);
    if (c == 0) gatomic_add_f32(cnt + edge, 1.0f);
}

// ---------------- per-edge mean (in place) ----------------
__global__ __launch_bounds__(256) void k_edge_mean(
    float* __restrict__ esum, const float* __restrict__ cnt) {
    int t = blockIdx.x * 256 + threadIdx.x;
    if (t >= NE * HO) return;
    float c = cnt[t >> 7];
    esum[t] = esum[t] / fmaxf(c, 1.0f);
}

// ---------------- node scatter: o[node] += 0.5 * emean[edge] ----------------
__global__ __launch_bounds__(256) void k_scatter_node(
    const int* __restrict__ idx, int nnz, const float* __restrict__ emean,
    float* __restrict__ o) {
    size_t t = (size_t)blockIdx.x * 256 + threadIdx.x;
    if (t >= (size_t)nnz * HO) return;
    int i = (int)(t >> 7);
    int c = (int)(t & 127);
    int node = idx[i];
    int edge = idx[nnz + i];
    gatomic_add_f32(o + (size_t)node * HO + c,
                    0.5f * emean[(size_t)edge * HO + c]);
}

// ---------------- GEMM2: out[10000,64] = o @ Wout^T + bout ----------------
__global__ __launch_bounds__(256) void k_gemm_out(
    const float* __restrict__ o, const bf16* __restrict__ w,
    const float* __restrict__ bias, float* __restrict__ out) {
    const int tile = blockIdx.x * 8 + (threadIdx.x >> 5);
    if (tile >= (NN / 16) * (OC / 16)) return;         // wave-uniform guard
    const int lane = threadIdx.x & 31;
    const int mt = tile / (OC / 16);
    const int nt = tile % (OC / 16);
    const int m0 = mt * 16, n0 = nt * 16;
    const int h = lane >> 4;
    const int l = lane & 15;
    const float* arow = o + (size_t)(m0 + l) * HO + h * 8;
    const bf16* brow = w + (size_t)(n0 + l) * HO + h * 8;
    v8f acc = {};
#pragma unroll
    for (int kb = 0; kb < HO; kb += 32) {
        v4f a0 = *(const v4f*)(arow + kb);
        v4f a1 = *(const v4f*)(arow + kb + 4);
        v4f a2 = *(const v4f*)(arow + kb + 16);
        v4f a3 = *(const v4f*)(arow + kb + 20);
        v8bf bl0 = *(const v8bf*)(brow + kb);
        v8bf bl1 = *(const v8bf*)(brow + kb + 16);
        v16bf a, b;
#pragma unroll
        for (int e = 0; e < 4; ++e) {
            a[e]      = (bf16)a0[e];
            a[4 + e]  = (bf16)a1[e];
            a[8 + e]  = (bf16)a2[e];
            a[12 + e] = (bf16)a3[e];
        }
#pragma unroll
        for (int e = 0; e < 8; ++e) { b[e] = bl0[e]; b[8 + e] = bl1[e]; }
        acc = __builtin_amdgcn_wmma_f32_16x16x32_bf16(
            false, a, false, b, (short)0, acc, false, false);
    }
    const float bv = bias[n0 + l];
#pragma unroll
    for (int r = 0; r < 8; ++r)
        out[(size_t)(m0 + h * 8 + r) * OC + n0 + l] = acc[r] + bv;
}

// ---------------- host launcher ----------------
extern "C" void kernel_launch(void* const* d_in, const int* in_sizes, int n_in,
                              void* d_out, int out_size, void* d_ws, size_t ws_size,
                              hipStream_t stream) {
    const float* x    = (const float*)d_in[0];
    const int*   hi1  = (const int*)d_in[1];
    const int*   hi2  = (const int*)d_in[2];
    const float* W1   = (const float*)d_in[3];
    const float* b1   = (const float*)d_in[4];
    const float* W2   = (const float*)d_in[5];
    const float* b2   = (const float*)d_in[6];
    const float* Wout = (const float*)d_in[7];
    const float* bout = (const float*)d_in[8];
    const int nnz1 = in_sizes[1] / 2;
    const int nnz2 = in_sizes[2] / 2;

    char* ws = (char*)d_ws;
    bf16*  xb   = (bf16*)(ws + OFF_XBF);
    bf16*  wcat = (bf16*)(ws + OFF_WCAT);
    float* bcat = (float*)(ws + OFF_BCAT);
    bf16*  wob  = (bf16*)(ws + OFF_WOUT);
    float* xt   = (float*)(ws + OFF_XT);
    float* es1  = (float*)(ws + OFF_ES1);
    float* cn1  = (float*)(ws + OFF_CN1);
    float* es2  = (float*)(ws + OFF_ES2);
    float* cn2  = (float*)(ws + OFF_CN2);
    float* oacc = (float*)(ws + OFF_O);
    float* out  = (float*)d_out;

    // 1) convert inputs to bf16 + concat weights/bias
    {
        int total = NN * KC + HO2 * KC + HO2 + OC * HO;
        k_convert<<<(total + 255) / 256, 256, 0, stream>>>(
            x, W1, b1, W2, b2, Wout, xb, wcat, bcat, wob);
    }
    // 2) zero accumulators (esum1,cnt1,esum2,cnt2,o are contiguous)
    k_zero<<<(unsigned)((ZERO_FLOATS + 255) / 256), 256, 0, stream>>>(es1, ZERO_FLOATS);
    // 3) fused projection GEMM for both scales (WMMA bf16 -> f32)
    k_gemm_xt<<<(NN / 16) * (HO2 / 16) / 8, 256, 0, stream>>>(xb, wcat, bcat, xt);
    // 4) scale 1: edge sum/count -> mean -> node scatter (0.5 folded in)
    {
        size_t t1 = (size_t)nnz1 * HO;
        k_scatter_edge<<<(unsigned)((t1 + 255) / 256), 256, 0, stream>>>(
            hi1, nnz1, xt, 0, es1, cn1);
        k_edge_mean<<<(NE * HO + 255) / 256, 256, 0, stream>>>(es1, cn1);
        k_scatter_node<<<(unsigned)((t1 + 255) / 256), 256, 0, stream>>>(
            hi1, nnz1, es1, oacc);
    }
    // 5) scale 2
    {
        size_t t2 = (size_t)nnz2 * HO;
        k_scatter_edge<<<(unsigned)((t2 + 255) / 256), 256, 0, stream>>>(
            hi2, nnz2, xt, HO, es2, cn2);
        k_edge_mean<<<(NE * HO + 255) / 256, 256, 0, stream>>>(es2, cn2);
        k_scatter_node<<<(unsigned)((t2 + 255) / 256), 256, 0, stream>>>(
            hi2, nnz2, es2, oacc);
    }
    // 6) output projection GEMM (WMMA bf16 -> f32), writes all of d_out
    k_gemm_out<<<((NN / 16) * (OC / 16) + 7) / 8, 256, 0, stream>>>(
        oacc, wob, bout, out);
}
